// TransformerEncoder_82832739270753
// MI455X (gfx1250) — matrix operand order repaired and checked by reading, not compiled, not run on hardware
//
#include <hip/hip_runtime.h>
#include <hip/hip_bf16.h>

typedef __attribute__((ext_vector_type(16))) _Float16 v16h;
typedef __attribute__((ext_vector_type(8)))  _Float16 v8h;
typedef __attribute__((ext_vector_type(8)))  float    v8f;
typedef __attribute__((ext_vector_type(4)))  float    v4f;

// ---------------------------------------------------------------------------
// GEMM (NT, fp32 B): C[z] = alpha * A[z] (MxK, lda) * B[z]^T (B is NxK rows)
//            + bias (+C if accumulate) (relu optional)
// K must be a multiple of 32. 8 waves/block -> 128x64 output tile.
// WMMA fragment layouts per CDNA5 ISA 7.12.2 (wave32).
// ---------------------------------------------------------------------------
__global__ __launch_bounds__(256) void gemm_nt_wmma(
    const float* __restrict__ A, const float* __restrict__ B,
    const float* __restrict__ bias, float* __restrict__ C,
    int M, int N, int K, int lda, int ldb, int ldc,
    long sA1, long sA2, long sB1, long sB2, long sC1, long sC2,
    int nz2, float alpha, int flags)
{
  const int z  = blockIdx.z;
  const int z1 = z / nz2, z2 = z - z1 * nz2;
  A += (long)z1 * sA1 + (long)z2 * sA2;
  B += (long)z1 * sB1 + (long)z2 * sB2;
  C += (long)z1 * sC1 + (long)z2 * sC2;

  const int wave = threadIdx.x >> 5;
  const int lane = threadIdx.x & 31;
  const int half = lane >> 4;      // which 16-lane half
  const int l15  = lane & 15;
  const int tileM = blockIdx.y * 128 + wave * 16;
  const int tileN = blockIdx.x * 64;
  if (tileM >= M) return;          // wave-uniform exit (no barriers in kernel)

  int ar = tileM + l15; if (ar >= M) ar = M - 1;    // clamp (stores guarded)
  const float* Ap = A + (long)ar * lda;

  const float* Bp[4];
#pragma unroll
  for (int j = 0; j < 4; ++j) {
    int n = tileN + j * 16 + l15; if (n >= N) n = N - 1;
    Bp[j] = B + (long)n * ldb;
  }

  v8f acc[4];
#pragma unroll
  for (int j = 0; j < 4; ++j) acc[j] = v8f{};

  for (int k0 = 0; k0 < K; k0 += 32) {
    // A fragment: lanes 0-15 hold K = k0+{0..7,16..23}; lanes 16-31 shift by 8.
    const int ka = k0 + half * 8;
    v4f a0 = *(const v4f*)(Ap + ka);
    v4f a1 = *(const v4f*)(Ap + ka + 4);
    v4f a2 = *(const v4f*)(Ap + ka + 16);
    v4f a3 = *(const v4f*)(Ap + ka + 20);
    if (k0 + 64 <= K) __builtin_prefetch(Ap + k0 + 32, 0, 1);
    v16h af;
#pragma unroll
    for (int i = 0; i < 4; ++i) {
      af[i]      = (_Float16)a0[i];
      af[4 + i]  = (_Float16)a1[i];
      af[8 + i]  = (_Float16)a2[i];
      af[12 + i] = (_Float16)a3[i];
    }
    // B fragment: same K layout as A (lanes 0-15 K = k0+{0..7,16..23}; 16-31 shift by 8).
    const int kb = k0 + half * 8;
#pragma unroll
    for (int j = 0; j < 4; ++j) {
      v4f b0 = *(const v4f*)(Bp[j] + kb);
      v4f b1 = *(const v4f*)(Bp[j] + kb + 4);
      v4f b2 = *(const v4f*)(Bp[j] + kb + 16);
      v4f b3 = *(const v4f*)(Bp[j] + kb + 20);
      v16h bf;
#pragma unroll
      for (int i = 0; i < 4; ++i) {
        bf[i]      = (_Float16)b0[i];
        bf[4 + i]  = (_Float16)b1[i];
        bf[8 + i]  = (_Float16)b2[i];
        bf[12 + i] = (_Float16)b3[i];
      }
      acc[j] = __builtin_amdgcn_wmma_f32_16x16x32_f16(
          false, af, false, bf, (short)0, acc[j], false, false);
    }
  }

#pragma unroll
  for (int j = 0; j < 4; ++j) {
    int n = tileN + j * 16 + l15;
    if (n < N) {
      float bv = bias ? bias[n] : 0.0f;
#pragma unroll
      for (int r = 0; r < 8; ++r) {
        int m = tileM + half * 8 + r;
        if (m < M) {
          long off = (long)m * ldc + n;
          float v = alpha * acc[j][r] + bv;
          if (flags & 2) v += C[off];
          if (flags & 1) v = fmaxf(v, 0.0f);
          C[off] = v;
        }
      }
    }
  }
}

// ---------------------------------------------------------------------------
// GEMM (NT, pre-converted f16 weights): C = alpha*A*Bh^T + bias (+C)(relu).
// Bh rows are K contiguous halves; fragment = two aligned b128 loads, zero
// per-iteration conversion on the weight side, half the L2 weight traffic.
// ---------------------------------------------------------------------------
__global__ __launch_bounds__(256) void gemm_nt_wmma_h(
    const float* __restrict__ A, const _Float16* __restrict__ B,
    const float* __restrict__ bias, float* __restrict__ C,
    int M, int N, int K, int lda, int ldb, int ldc,
    long sA1, long sA2, long sB1, long sB2, long sC1, long sC2,
    int nz2, float alpha, int flags)
{
  const int z  = blockIdx.z;
  const int z1 = z / nz2, z2 = z - z1 * nz2;
  A += (long)z1 * sA1 + (long)z2 * sA2;
  B += (long)z1 * sB1 + (long)z2 * sB2;
  C += (long)z1 * sC1 + (long)z2 * sC2;

  const int wave = threadIdx.x >> 5;
  const int lane = threadIdx.x & 31;
  const int half = lane >> 4;
  const int l15  = lane & 15;
  const int tileM = blockIdx.y * 128 + wave * 16;
  const int tileN = blockIdx.x * 64;
  if (tileM >= M) return;

  int ar = tileM + l15; if (ar >= M) ar = M - 1;
  const float* Ap = A + (long)ar * lda;

  const _Float16* Bp[4];
#pragma unroll
  for (int j = 0; j < 4; ++j) {
    int n = tileN + j * 16 + l15; if (n >= N) n = N - 1;
    Bp[j] = B + (long)n * ldb;
  }

  v8f acc[4];
#pragma unroll
  for (int j = 0; j < 4; ++j) acc[j] = v8f{};

  for (int k0 = 0; k0 < K; k0 += 32) {
    const int ka = k0 + half * 8;
    v4f a0 = *(const v4f*)(Ap + ka);
    v4f a1 = *(const v4f*)(Ap + ka + 4);
    v4f a2 = *(const v4f*)(Ap + ka + 16);
    v4f a3 = *(const v4f*)(Ap + ka + 20);
    if (k0 + 64 <= K) __builtin_prefetch(Ap + k0 + 32, 0, 1);
    v16h af;
#pragma unroll
    for (int i = 0; i < 4; ++i) {
      af[i]      = (_Float16)a0[i];
      af[4 + i]  = (_Float16)a1[i];
      af[8 + i]  = (_Float16)a2[i];
      af[12 + i] = (_Float16)a3[i];
    }
    const int kb = k0 + half * 8;
#pragma unroll
    for (int j = 0; j < 4; ++j) {
      v8h h0 = *(const v8h*)(Bp[j] + kb);
      v8h h1 = *(const v8h*)(Bp[j] + kb + 16);
      v16h bf;
#pragma unroll
      for (int i = 0; i < 8; ++i) { bf[i] = h0[i]; bf[8 + i] = h1[i]; }
      acc[j] = __builtin_amdgcn_wmma_f32_16x16x32_f16(
          false, af, false, bf, (short)0, acc[j], false, false);
    }
  }

#pragma unroll
  for (int j = 0; j < 4; ++j) {
    int n = tileN + j * 16 + l15;
    if (n < N) {
      float bv = bias ? bias[n] : 0.0f;
#pragma unroll
      for (int r = 0; r < 8; ++r) {
        int m = tileM + half * 8 + r;
        if (m < M) {
          long off = (long)m * ldc + n;
          float v = alpha * acc[j][r] + bv;
          if (flags & 2) v += C[off];
          if (flags & 1) v = fmaxf(v, 0.0f);
          C[off] = v;
        }
      }
    }
  }
}

// ---------------------------------------------------------------------------
// GEMM (NN): C[z] = alpha * A[z] (MxK, lda) * B[z] (KxN, ldb) (+bias)(+C)
// Arbitrary K (zero-padded tail).  B tiles staged through LDS as f16
// (stride 40 halves => aligned ds_load_b128 fragment reads).
// ---------------------------------------------------------------------------
__global__ __launch_bounds__(256) void gemm_nn_wmma(
    const float* __restrict__ A, const float* __restrict__ B,
    const float* __restrict__ bias, float* __restrict__ C,
    int M, int N, int K, int lda, int ldb, int ldc,
    long sA1, long sA2, long sB1, long sB2, long sC1, long sC2,
    int nz2, float alpha, int flags)
{
  __shared__ _Float16 sB[64 * 40];

  const int z  = blockIdx.z;
  const int z1 = z / nz2, z2 = z - z1 * nz2;
  A += (long)z1 * sA1 + (long)z2 * sA2;
  B += (long)z1 * sB1 + (long)z2 * sB2;
  C += (long)z1 * sC1 + (long)z2 * sC2;

  const int wave = threadIdx.x >> 5;
  const int lane = threadIdx.x & 31;
  const int half = lane >> 4;
  const int l15  = lane & 15;
  const int tileM = blockIdx.y * 128 + wave * 16;
  const int tileN = blockIdx.x * 64;

  int ar = tileM + l15; if (ar >= M) ar = M - 1;
  const float* Ap = A + (long)ar * lda;

  v8f acc[4];
#pragma unroll
  for (int j = 0; j < 4; ++j) acc[j] = v8f{};

  for (int k0 = 0; k0 < K; k0 += 32) {
    __syncthreads();
    // Cooperative stage of B[k0..k0+31][tileN..tileN+63] (coalesced over n).
#pragma unroll
    for (int i = 0; i < 8; ++i) {
      int idx = threadIdx.x + i * 256;
      int kk = idx >> 6, nn = idx & 63;
      int gk = k0 + kk, gn = tileN + nn;
      float v = (gk < K && gn < N) ? B[(long)gk * ldb + gn] : 0.0f;
      sB[nn * 40 + kk] = (_Float16)v;
    }
    __syncthreads();

    const int ka = k0 + half * 8;
    v16h af;
#pragma unroll
    for (int i = 0; i < 8; ++i) {
      int k1 = ka + i, k2 = ka + 16 + i;
      af[i]     = (_Float16)((k1 < K) ? Ap[k1] : 0.0f);
      af[8 + i] = (_Float16)((k2 < K) ? Ap[k2] : 0.0f);
    }
    const int kb = half * 8;
#pragma unroll
    for (int j = 0; j < 4; ++j) {
      const _Float16* bp = &sB[(j * 16 + l15) * 40 + kb];
      v8h h0 = *(const v8h*)bp;
      v8h h1 = *(const v8h*)(bp + 16);
      v16h bf;
#pragma unroll
      for (int i = 0; i < 8; ++i) { bf[i] = h0[i]; bf[8 + i] = h1[i]; }
      acc[j] = __builtin_amdgcn_wmma_f32_16x16x32_f16(
          false, af, false, bf, (short)0, acc[j], false, false);
    }
  }

#pragma unroll
  for (int j = 0; j < 4; ++j) {
    int n = tileN + j * 16 + l15;
    if (n < N) {
      float bv = bias ? bias[n] : 0.0f;
#pragma unroll
      for (int r = 0; r < 8; ++r) {
        int m = tileM + half * 8 + r;
        if (m < M) {
          long off = (long)m * ldc + n;
          float v = alpha * acc[j][r] + bv;
          if (flags & 2) v += C[off];
          if (flags & 1) v = fmaxf(v, 0.0f);
          C[off] = v;
        }
      }
    }
  }
}

// fp32 -> f16 weight conversion (one pass per launch; amortized by reuse)
__global__ void f32_to_f16_k(const float* __restrict__ X, _Float16* __restrict__ Y, long n)
{
  long i = (long)blockIdx.x * 256 + threadIdx.x;
  if (i < n) Y[i] = (_Float16)X[i];
}

// ---------------------------------------------------------------------------
// Row softmax (in place). grid = (rows_per_batch, batches); rows are packed
// with stride == cols inside each batch region of stride bstride.
// ---------------------------------------------------------------------------
__global__ __launch_bounds__(128) void softmax_rows_k(float* X, int cols, long bstride)
{
  float* x = X + (long)blockIdx.y * bstride + (long)blockIdx.x * cols;
  const int t = threadIdx.x;
  __shared__ float red[128];
  float mx = -3.4e38f;
  for (int c = t; c < cols; c += 128) mx = fmaxf(mx, x[c]);
  red[t] = mx; __syncthreads();
  for (int s = 64; s > 0; s >>= 1) { if (t < s) red[t] = fmaxf(red[t], red[t + s]); __syncthreads(); }
  mx = red[0]; __syncthreads();
  float sm = 0.0f;
  for (int c = t; c < cols; c += 128) { float e = __expf(x[c] - mx); x[c] = e; sm += e; }
  red[t] = sm; __syncthreads();
  for (int s = 64; s > 0; s >>= 1) { if (t < s) red[t] += red[t + s]; __syncthreads(); }
  float inv = 1.0f / red[0];
  for (int c = t; c < cols; c += 128) x[c] *= inv;
}

// Row L2-normalize: grid = (rows_per_batch, batches)
__global__ __launch_bounds__(128) void norm1_rows_k(float* X, int d, long rstride, long bstride)
{
  float* x = X + (long)blockIdx.y * bstride + (long)blockIdx.x * rstride;
  const int t = threadIdx.x;
  __shared__ float red[128];
  float ss = 0.0f;
  for (int c = t; c < d; c += 128) ss += x[c] * x[c];
  red[t] = ss; __syncthreads();
  for (int s = 64; s > 0; s >>= 1) { if (t < s) red[t] += red[t + s]; __syncthreads(); }
  float inv = 1.0f / fmaxf(sqrtf(red[0]), 1e-12f);
  for (int c = t; c < d; c += 128) x[c] *= inv;
}

// OUT[row] = LayerNorm(OUT[row] + R[row]) * s + b   (256 cols, block 256)
__global__ __launch_bounds__(256) void add_ln_k(float* O, const float* __restrict__ R,
                                                const float* __restrict__ s,
                                                const float* __restrict__ b)
{
  const long row = blockIdx.x;
  const int t = threadIdx.x;
  float v = O[row * 256 + t] + R[row * 256 + t];
  __shared__ float red[256];
  red[t] = v; __syncthreads();
  for (int st = 128; st > 0; st >>= 1) { if (t < st) red[t] += red[t + st]; __syncthreads(); }
  float mean = red[0] * (1.0f / 256.0f);
  __syncthreads();
  float d = v - mean;
  red[t] = d * d; __syncthreads();
  for (int st = 128; st > 0; st >>= 1) { if (t < st) red[t] += red[t + st]; __syncthreads(); }
  float var = red[0] * (1.0f / 256.0f);
  O[row * 256 + t] = d * rsqrtf(var + 1e-5f) * s[t] + b[t];
}

// (S,B,E) -> (B,S,E)
__global__ void sbe_to_bse_k(const float* __restrict__ X, float* __restrict__ Y, long total)
{
  long i = (long)blockIdx.x * 256 + threadIdx.x;
  if (i >= total) return;
  int e = i & 255; long t = i >> 8;
  int s = (int)(t % 196); int b = (int)(t / 196);
  Y[i] = X[((long)s * 128 + b) * 256 + e];
}
// (B,S,E) -> (S,B,E)
__global__ void bse_to_sbe_k(const float* __restrict__ X, float* __restrict__ Y, long total)
{
  long i = (long)blockIdx.x * 256 + threadIdx.x;
  if (i >= total) return;
  int e = i & 255; long t = i >> 8;
  int s = (int)(t % 196); int b = (int)(t / 196);
  Y[((long)s * 128 + b) * 256 + e] = X[i];
}

__global__ void add2_k(const float* __restrict__ X, const float* __restrict__ Y,
                       float* __restrict__ Z, long total)
{
  long i = (long)blockIdx.x * 256 + threadIdx.x;
  if (i < total) Z[i] = X[i] + Y[i];
}

// Patch embed (9 or 4 inputs -> 256, then row L2 norm). grid = B*196, block 256
__global__ __launch_bounds__(256) void embed256_k(const float* __restrict__ X, int off, int Kin,
                                                  const float* __restrict__ W,
                                                  const float* __restrict__ bias,
                                                  float* __restrict__ Y)
{
  const long p = blockIdx.x; const int t = threadIdx.x;
  const float* x = X + p * 13 + off;
  float acc = bias[t];
  for (int k = 0; k < Kin; ++k) acc += x[k] * W[t * Kin + k];
  __shared__ float red[256];
  red[t] = acc * acc; __syncthreads();
  for (int st = 128; st > 0; st >>= 1) { if (t < st) red[t] += red[t + st]; __syncthreads(); }
  float inv = 1.0f / fmaxf(sqrtf(red[0]), 1e-12f);
  Y[p * 256 + t] = acc * inv;
}

// Canny embed: 4 slices of 16 -> 128, L2 norm per slice, into cf (B,196,512)
__global__ __launch_bounds__(128) void embed_tdlr_k(const float* __restrict__ X,
                                                    const float* __restrict__ W,
                                                    const float* __restrict__ bias,
                                                    float* __restrict__ cf)
{
  const long p = blockIdx.x; const int s = blockIdx.y; const int t = threadIdx.x;
  const float* x = X + p * 64 + s * 16;
  float acc = bias[t];
#pragma unroll
  for (int k = 0; k < 16; ++k) acc += x[k] * W[t * 16 + k];
  __shared__ float red[128];
  red[t] = acc * acc; __syncthreads();
  for (int st = 64; st > 0; st >>= 1) { if (t < st) red[t] += red[t + st]; __syncthreads(); }
  float inv = 1.0f / fmaxf(sqrtf(red[0]), 1e-12f);
  cf[p * 512 + s * 128 + t] = acc * inv;
}

// Column softmax of pool^T: Sm[b,m,n] = softmax_m pool[b,n,m]. grid=(n,B), block 128
__global__ __launch_bounds__(128) void pool_sm_k(const float* __restrict__ pool,
                                                 float* __restrict__ Sm, int n, int m)
{
  const int b = blockIdx.y, col = blockIdx.x, t = threadIdx.x;
  const float* x = pool + ((long)b * n + col) * m;
  __shared__ float red[128];
  float v = (t < m) ? x[t] : -3.4e38f;
  red[t] = v; __syncthreads();
  for (int st = 64; st > 0; st >>= 1) { if (t < st) red[t] = fmaxf(red[t], red[t + st]); __syncthreads(); }
  float mx = red[0]; __syncthreads();
  float e = (t < m) ? __expf(v - mx) : 0.0f;
  red[t] = e; __syncthreads();
  for (int st = 64; st > 0; st >>= 1) { if (t < st) red[t] += red[t + st]; __syncthreads(); }
  if (t < m) Sm[((long)b * m + t) * n + col] = e / red[0];
}

// Canny 2x2 similarity max (2-way softmax == sigmoid of dot difference)
__global__ __launch_bounds__(128) void canny_sim_k(const float* __restrict__ cf,
                                                   float* __restrict__ mbuf, int n)
{
  const float* base = cf + ((long)blockIdx.y * 196 + blockIdx.x) * 512;
  const int t = threadIdx.x;
  float ct = base[t], cd = base[128 + t], cl = base[256 + t], cr = base[384 + t];
  __shared__ float red[6][128];
  red[0][t] = ct * ct; red[1][t] = ct * cd; red[2][t] = cd * cd;
  red[3][t] = cl * cl; red[4][t] = cl * cr; red[5][t] = cr * cr;
  __syncthreads();
  for (int st = 64; st > 0; st >>= 1) {
    if (t < st)
#pragma unroll
      for (int q = 0; q < 6; ++q) red[q][t] += red[q][t + st];
    __syncthreads();
  }
  if (t == 0) {
    float s1 = 1.0f / (1.0f + __expf(red[0][0] - red[1][0]));  // td_sim[0,1]
    float s2 = 1.0f / (1.0f + __expf(red[2][0] - red[1][0]));  // td_sim[1,0]
    float s3 = 1.0f / (1.0f + __expf(red[3][0] - red[4][0]));  // lr_sim[0,1]
    float s4 = 1.0f / (1.0f + __expf(red[5][0] - red[4][0]));  // lr_sim[1,0]
    mbuf[blockIdx.y * 196 + blockIdx.x] = fmaxf(fmaxf(s1, s2), fmaxf(s3, s4));
  }
}

// row[b,:] = sum_n m[b,n] * cf[b,n,:]   (512 dims, block 256)
__global__ __launch_bounds__(256) void canny_row_k(const float* __restrict__ cf,
                                                   const float* __restrict__ mbuf,
                                                   float* __restrict__ row, int n)
{
  const int b = blockIdx.x, t = threadIdx.x;
  float a0 = 0.0f, a1 = 0.0f;
  for (int r = 0; r < n; ++r) {
    float m = mbuf[b * 196 + r];
    const float* p = cf + ((long)b * 196 + r) * 512;
    a0 += m * p[t]; a1 += m * p[t + 256];
  }
  row[b * 512 + t] = a0; row[b * 512 + t + 256] = a1;
}

// cf[b, r<mrows, :] = row[b,:] / ||row[b]||  (pool+norm1 of a broadcast row is
// exactly the normalized row since softmax weights are positive)
__global__ __launch_bounds__(256) void canny_bcast_k(const float* __restrict__ row,
                                                     float* __restrict__ cf, int mrows)
{
  const int b = blockIdx.x, t = threadIdx.x;
  float v0 = row[b * 512 + t], v1 = row[b * 512 + t + 256];
  __shared__ float red[256];
  red[t] = v0 * v0 + v1 * v1; __syncthreads();
  for (int st = 128; st > 0; st >>= 1) { if (t < st) red[t] += red[t + st]; __syncthreads(); }
  float inv = 1.0f / fmaxf(sqrtf(red[0]), 1e-12f);
  v0 *= inv; v1 *= inv;
  for (int r = 0; r < mrows; ++r) {
    float* p = cf + ((long)b * 196 + r) * 512;
    p[t] = v0; p[t + 256] = v1;
  }
}

// pcat[b, 0:m]=wco*pc, [m:2m]=wpo*pp, [2m:3m]=wc*cfeat
__global__ void pcat_build_k(const float* __restrict__ pc, const float* __restrict__ pp,
                             const float* __restrict__ cfeat,
                             const float* __restrict__ wco, const float* __restrict__ wpo,
                             const float* __restrict__ wc,
                             float* __restrict__ pcat, int m, long total)
{
  long i = (long)blockIdx.x * 256 + threadIdx.x;
  if (i >= total) return;
  int e = i & 255; long t = i >> 8;
  int tt = (int)(t % (3 * m)); int b = (int)(t / (3 * m));
  int seg = tt / m, r = tt - seg * m;
  const float* src = (seg == 0) ? pc : (seg == 1) ? pp : cfeat;
  float w = (seg == 0) ? wco[0] : (seg == 1) ? wpo[0] : wc[0];
  pcat[i] = w * src[((long)b * 196 + r) * 256 + e];
}

// ---------------------------------------------------------------------------
extern "C" void kernel_launch(void* const* d_in, const int* in_sizes, int n_in,
                              void* d_out, int out_size, void* d_ws, size_t ws_size,
                              hipStream_t stream)
{
  (void)in_sizes; (void)n_in; (void)out_size; (void)ws_size;
  const float* src     = (const float*)d_in[0];
  const float* patch   = (const float*)d_in[1];
  const float* canny   = (const float*)d_in[2];
  const float* pos     = (const float*)d_in[3];
  const float* color_w = (const float*)d_in[4];
  const float* color_b = (const float*)d_in[5];
  const float* posp_w  = (const float*)d_in[6];
  const float* posp_b  = (const float*)d_in[7];
  const float* tdlr_w  = (const float*)d_in[8];
  const float* tdlr_b  = (const float*)d_in[9];
  const float* cproj_w = (const float*)d_in[10];
  const float* cproj_b = (const float*)d_in[11];
  const float* a0_in_w = (const float*)d_in[12];
  const float* a0_in_b = (const float*)d_in[13];
  const float* a0_out_w= (const float*)d_in[14];
  const float* a0_out_b= (const float*)d_in[15];
  const float* wco     = (const float*)d_in[16];
  const float* wpo     = (const float*)d_in[17];
  const float* wc      = (const float*)d_in[18];
  const float* l_in_w  = (const float*)d_in[19];
  const float* l_in_b  = (const float*)d_in[20];
  const float* l_out_w = (const float*)d_in[21];
  const float* l_out_b = (const float*)d_in[22];
  const float* f1_w    = (const float*)d_in[23];
  const float* f1_b    = (const float*)d_in[24];
  const float* f2_w    = (const float*)d_in[25];
  const float* f2_b    = (const float*)d_in[26];
  const float* n1_s    = (const float*)d_in[27];
  const float* n1_b    = (const float*)d_in[28];
  const float* n2_s    = (const float*)d_in[29];
  const float* n2_b    = (const float*)d_in[30];
  const float* pools[6];
  for (int i = 0; i < 6; ++i) pools[i] = (const float*)d_in[34 + i];
  float* OUTF = (float*)d_out;

  const int B = 128, S = 196, E = 256;
  const int narr[6] = {196, 120, 100, 80, 70, 60};
  const int marr[6] = {120, 100, 80, 70, 60, 50};
  const long tot = (long)B * S * E;

  // workspace carve-up
  float* wsf = (float*)d_ws;
  size_t o = 0;
  auto alloc = [&](size_t n) { float* p = wsf + o; o += n; return p; };
  float* OUT   = alloc(tot);
  float* POS   = alloc(tot);
  float* QK    = alloc(tot);
  float* QKV   = alloc((size_t)B * 360 * 768);
  float* SC    = alloc((size_t)B * 4 * 360 * 360);
  float* TMP   = alloc((size_t)B * 360 * 256);
  float* PCAT  = alloc((size_t)B * 360 * 256);
  float* PCAT2 = alloc((size_t)B * 360 * 256);
  float* SRC2  = alloc(tot);
  float* FFH   = alloc((size_t)B * S * 2048);
  float* PCa   = alloc((size_t)B * 196 * 256);
  float* PCb   = alloc((size_t)B * 196 * 256);
  float* PPa   = alloc((size_t)B * 196 * 256);
  float* PPb   = alloc((size_t)B * 196 * 256);
  float* CF    = alloc((size_t)B * 196 * 512);
  float* CFEAT = alloc((size_t)B * 196 * 256);
  float* SMS[6];
  for (int i = 0; i < 6; ++i) SMS[i] = alloc((size_t)B * marr[i] * narr[i]);
  float* MBUF  = alloc((size_t)B * 196);
  float* ROW   = alloc((size_t)B * 512);

  // f16 weight cache (converted once per launch)
  const long nA0IN = 768L * 256, nA0OUT = 256L * 256, nLIN = 6L * 768 * 256,
             nLOUT = 6L * 256 * 256, nF1 = 6L * 2048 * 256, nF2 = 6L * 256 * 2048,
             nCPROJ = 256L * 512;
  const long nH = nA0IN + nA0OUT + nLIN + nLOUT + nF1 + nF2 + nCPROJ;
  _Float16* WH = (_Float16*)alloc((size_t)(nH / 2 + 16));
  _Float16* hA0IN  = WH;
  _Float16* hA0OUT = hA0IN  + nA0IN;
  _Float16* hLIN   = hA0OUT + nA0OUT;
  _Float16* hLOUT  = hLIN   + nLIN;
  _Float16* hF1    = hLOUT  + nLOUT;
  _Float16* hF2    = hF1    + nF1;
  _Float16* hCPROJ = hF2    + nF2;

  auto cvt = [&](const float* X, _Float16* Y, long n) {
    f32_to_f16_k<<<(unsigned)((n + 255) / 256), 256, 0, stream>>>(X, Y, n);
  };
  cvt(a0_in_w,  hA0IN,  nA0IN);
  cvt(a0_out_w, hA0OUT, nA0OUT);
  cvt(l_in_w,   hLIN,   nLIN);
  cvt(l_out_w,  hLOUT,  nLOUT);
  cvt(f1_w,     hF1,    nF1);
  cvt(f2_w,     hF2,    nF2);
  cvt(cproj_w,  hCPROJ, nCPROJ);

  auto gemmNT = [&](const float* A, const float* Bm, const float* bias, float* C,
                    int M, int N, int K, int lda, int ldb, int ldc,
                    long sA1, long sA2, long sB1, long sB2, long sC1, long sC2,
                    int nz1, int nz2, float alpha, int flags) {
    dim3 g((unsigned)((N + 63) / 64), (unsigned)((M + 127) / 128), (unsigned)(nz1 * nz2));
    gemm_nt_wmma<<<g, 256, 0, stream>>>(A, Bm, bias, C, M, N, K, lda, ldb, ldc,
                                        sA1, sA2, sB1, sB2, sC1, sC2, nz2, alpha, flags);
  };
  auto gemmNTH = [&](const float* A, const _Float16* Bm, const float* bias, float* C,
                     int M, int N, int K, int lda, int ldb, int ldc,
                     float alpha, int flags) {
    dim3 g((unsigned)((N + 63) / 64), (unsigned)((M + 127) / 128), 1u);
    gemm_nt_wmma_h<<<g, 256, 0, stream>>>(A, Bm, bias, C, M, N, K, lda, ldb, ldc,
                                          0, 0, 0, 0, 0, 0, 1, alpha, flags);
  };
  auto gemmNTHb = [&](const float* A, const _Float16* Bm, const float* bias, float* C,
                      int M, int N, int K, int lda, int ldb, int ldc,
                      long sA1, long sC1, int nz1, float alpha, int flags) {
    dim3 g((unsigned)((N + 63) / 64), (unsigned)((M + 127) / 128), (unsigned)nz1);
    gemm_nt_wmma_h<<<g, 256, 0, stream>>>(A, Bm, bias, C, M, N, K, lda, ldb, ldc,
                                          sA1, 0, 0, 0, sC1, 0, 1, alpha, flags);
  };
  auto gemmNN = [&](const float* A, const float* Bm, const float* bias, float* C,
                    int M, int N, int K, int lda, int ldb, int ldc,
                    long sA1, long sA2, long sB1, long sB2, long sC1, long sC2,
                    int nz1, int nz2, float alpha, int flags) {
    dim3 g((unsigned)((N + 63) / 64), (unsigned)((M + 127) / 128), (unsigned)(nz1 * nz2));
    gemm_nn_wmma<<<g, 256, 0, stream>>>(A, Bm, bias, C, M, N, K, lda, ldb, ldc,
                                        sA1, sA2, sB1, sB2, sC1, sC2, nz2, alpha, flags);
  };
  auto softm = [&](float* X, int rows, int cols, int batches, long bstride) {
    softmax_rows_k<<<dim3((unsigned)rows, (unsigned)batches), 128, 0, stream>>>(X, cols, bstride);
  };

  // ---- prologue: layout transposes, embeds, pool softmaxes -----------------
  sbe_to_bse_k<<<(unsigned)((tot + 255) / 256), 256, 0, stream>>>(src, OUT, tot);
  sbe_to_bse_k<<<(unsigned)((tot + 255) / 256), 256, 0, stream>>>(pos, POS, tot);
  embed256_k<<<B * 196, 256, 0, stream>>>(patch, 0, 9, color_w, color_b, PCa);
  embed256_k<<<B * 196, 256, 0, stream>>>(patch, 9, 4, posp_w, posp_b, PPa);
  embed_tdlr_k<<<dim3(B * 196, 4), 128, 0, stream>>>(canny, tdlr_w, tdlr_b, CF);
  for (int i = 0; i < 6; ++i)
    pool_sm_k<<<dim3((unsigned)narr[i], B), 128, 0, stream>>>(pools[i], SMS[i], narr[i], marr[i]);

  float* pcCur = PCa; float* pcNxt = PCb;
  float* ppCur = PPa; float* ppNxt = PPb;

  for (int i = 0; i < 6; ++i) {
    const int n = narr[i], m = marr[i], T = 3 * m;

    // ---- pc branch: Wco = softmax(pc pc^T); pc = norm1(Sm (Wco pc)) --------
    gemmNT(pcCur, pcCur, nullptr, SC, n, n, 256, 256, 256, n,
           196L * 256, 0, 196L * 256, 0, 196L * 196, 0, B, 1, 1.0f, 0);
    softm(SC, n, n, B, 196L * 196);
    gemmNN(SC, pcCur, nullptr, TMP, n, 256, n, n, 256, 256,
           196L * 196, 0, 196L * 256, 0, 196L * 256, 0, B, 1, 1.0f, 0);
    gemmNN(SMS[i], TMP, nullptr, pcNxt, m, 256, n, n, 256, 256,
           (long)m * n, 0, 196L * 256, 0, 196L * 256, 0, B, 1, 1.0f, 0);
    norm1_rows_k<<<dim3((unsigned)m, B), 128, 0, stream>>>(pcNxt, 256, 256, 196L * 256);

    // ---- pp branch ---------------------------------------------------------
    gemmNT(ppCur, ppCur, nullptr, SC, n, n, 256, 256, 256, n,
           196L * 256, 0, 196L * 256, 0, 196L * 196, 0, B, 1, 1.0f, 0);
    softm(SC, n, n, B, 196L * 196);
    gemmNN(SC, ppCur, nullptr, TMP, n, 256, n, n, 256, 256,
           196L * 196, 0, 196L * 256, 0, 196L * 256, 0, B, 1, 1.0f, 0);
    gemmNN(SMS[i], TMP, nullptr, ppNxt, m, 256, n, n, 256, 256,
           (long)m * n, 0, 196L * 256, 0, 196L * 256, 0, B, 1, 1.0f, 0);
    norm1_rows_k<<<dim3((unsigned)m, B), 128, 0, stream>>>(ppNxt, 256, 256, 196L * 256);

    // ---- canny branch ------------------------------------------------------
    canny_sim_k<<<dim3((unsigned)n, B), 128, 0, stream>>>(CF, MBUF, n);
    canny_row_k<<<B, 256, 0, stream>>>(CF, MBUF, ROW, n);
    canny_bcast_k<<<B, 256, 0, stream>>>(ROW, CF, m);
    gemmNTHb(CF, hCPROJ, cproj_b, CFEAT, m, 256, 512, 512, 512, 256,
             196L * 512, 196L * 256, B, 1.0f, 0);

    // ---- pcat + auxiliary MHA (nh=4, hd=64) --------------------------------
    long ptot = (long)B * T * 256;
    pcat_build_k<<<(unsigned)((ptot + 255) / 256), 256, 0, stream>>>(
        pcNxt, ppNxt, CFEAT, wco, wpo, wc, PCAT, m, ptot);
    gemmNTH(PCAT, hA0IN, a0_in_b, QKV, B * T, 768, 256, 256, 256, 768, 1.0f, 0);
    gemmNT(QKV, QKV + 256, nullptr, SC, T, T, 64, 768, 768, T,
           (long)T * 768, 64, (long)T * 768, 64, 4L * T * T, (long)T * T, B, 4, 0.125f, 0);
    softm(SC, T, T, B * 4, (long)T * T);
    gemmNN(SC, QKV + 512, nullptr, TMP, T, 64, T, T, 768, 256,
           4L * T * T, (long)T * T, (long)T * 768, 64, (long)T * 256, 64, B, 4, 1.0f, 0);
    gemmNTH(TMP, hA0OUT, a0_out_b, PCAT2, B * T, 256, 256, 256, 256, 256, 1.0f, 0);

    // ---- cross attention: output += softmax(output pcat^T) pcat ------------
    gemmNT(OUT, PCAT2, nullptr, SC, S, T, 256, 256, 256, T,
           (long)S * E, 0, (long)T * 256, 0, (long)S * T, 0, B, 1, 1.0f, 0);
    softm(SC, S, T, B, (long)S * T);
    gemmNN(SC, PCAT2, nullptr, OUT, S, 256, T, T, 256, 256,
           (long)S * T, 0, (long)T * 256, 0, (long)S * E, 0, B, 1, 1.0f, 2);

    // ---- layer MHA (nh=8, hd=32): q,k from output+pos; v from output -------
    add2_k<<<(unsigned)((tot + 255) / 256), 256, 0, stream>>>(OUT, POS, QK, tot);
    const _Float16* liwH = hLIN + (long)i * 768 * 256;
    const float* lib = l_in_b + (long)i * 768;
    gemmNTH(QK, liwH, lib, QKV, B * S, 512, 256, 256, 256, 768, 1.0f, 0);
    gemmNTH(OUT, liwH + 512L * 256, lib + 512, QKV + 512, B * S, 256, 256,
            256, 256, 768, 1.0f, 0);
    gemmNT(QKV, QKV + 256, nullptr, SC, S, S, 32, 768, 768, S,
           (long)S * 768, 32, (long)S * 768, 32, 8L * S * S, (long)S * S, B, 8,
           0.1767766953f, 0);
    softm(SC, S, S, B * 8, (long)S * S);
    gemmNN(SC, QKV + 512, nullptr, TMP, S, 32, S, S, 768, 256,
           8L * S * S, (long)S * S, (long)S * 768, 32, (long)S * E, 32, B, 8, 1.0f, 0);
    gemmNTH(TMP, hLOUT + (long)i * 256 * 256, l_out_b + (long)i * 256, SRC2,
            B * S, 256, 256, 256, 256, 256, 1.0f, 0);
    add_ln_k<<<B * S, 256, 0, stream>>>(OUT, SRC2, n1_s + (long)i * 256, n1_b + (long)i * 256);

    // ---- FFN ---------------------------------------------------------------
    gemmNTH(OUT, hF1 + (long)i * 2048 * 256, f1_b + (long)i * 2048, FFH,
            B * S, 2048, 256, 256, 256, 2048, 1.0f, 1);
    gemmNTH(FFH, hF2 + (long)i * 256 * 2048, f2_b + (long)i * 256, SRC2,
            B * S, 256, 2048, 256, 2048, 256, 1.0f, 0);
    add_ln_k<<<B * S, 256, 0, stream>>>(OUT, SRC2, n2_s + (long)i * 256, n2_b + (long)i * 256);

    { float* t0 = pcCur; pcCur = pcNxt; pcNxt = t0; }
    { float* t1 = ppCur; ppCur = ppNxt; ppNxt = t1; }
  }

  bse_to_sbe_k<<<(unsigned)((tot + 255) / 256), 256, 0, stream>>>(OUT, OUTF, tot);
}
